// TensorParallelMultiheadAttention_28338194219059
// MI455X (gfx1250) — matrix operand order repaired and checked by reading, not compiled, not run on hardware
//
#include <hip/hip_runtime.h>
#include <hip/hip_bf16.h>

typedef __bf16 bf16;
typedef __attribute__((ext_vector_type(16))) __bf16 v16bf;
typedef __attribute__((ext_vector_type(8)))  __bf16 v8bf;
typedef __attribute__((ext_vector_type(8)))  float  v8f;

#define BB    2
#define SQ    2048
#define H     1024
#define NH    16
#define HN    64
#define NATT  (BB*NH)     // 32
#define TOK   (SQ*BB)     // 4096

__device__ __forceinline__ v16bf cat8(v8bf lo, v8bf hi) {
  v16bf r;
#pragma unroll
  for (int i = 0; i < 8; ++i) { r[i] = lo[i]; r[i + 8] = hi[i]; }
  return r;
}
__device__ __forceinline__ v8bf ld8(const bf16* p) { return *(const v8bf*)p; }

// A fragment (16x32, bf16): lane row = lane&15; elems 0..7 at K = kofs + 8*hi,
// elems 8..15 at K = kofs + 16 + 8*hi   (hi = lane>=16)
__device__ __forceinline__ v16bf load_afrag(const bf16* rowptr, int hi) {
  const bf16* p = rowptr + hi * 8;
  return cat8(ld8(p), ld8(p + 16));
}
// B fragment (32x16, bf16): lane col = lane&15; elems 0..7 at K = kofs + 8*hi,
// elems 8..15 at K = kofs + 16 + 8*hi   (hi = lane>=16)
__device__ __forceinline__ v16bf load_bfrag(const bf16* colptr, int hi) {
  const bf16* p = colptr + hi * 8;
  return cat8(ld8(p), ld8(p + 16));
}
__device__ __forceinline__ v8f wmma_bf16(v16bf a, v16bf b, v8f c) {
  return __builtin_amdgcn_wmma_f32_16x16x32_bf16(false, a, false, b, (short)0, c, false, false);
}
// gfx1250 async global->LDS copy, 16 bytes per lane (tracked by ASYNCcnt)
__device__ __forceinline__ void async_cp16(const bf16* gsrc, bf16* ldst) {
  unsigned lofs = (unsigned)(size_t)ldst;      // low 32 bits of shared aperture = LDS offset
  asm volatile("global_load_async_to_lds_b128 %0, %1, off"
               :: "v"(lofs), "v"(gsrc) : "memory");
}

// ---------------- stage 1: convert to bf16 ----------------
__global__ __launch_bounds__(256) void cvt_x_kernel(const float* __restrict__ q, bf16* __restrict__ xb) {
  int i = blockIdx.x * 256 + threadIdx.x;          // over TOK*H
  int t = i >> 10, k = i & (H - 1);
  int s = t >> 1, bb = t & 1;
  xb[i] = (bf16)q[(bb * SQ + s) * H + k];
}
__global__ __launch_bounds__(256) void cvt_w_kernel(const float* __restrict__ w, bf16* __restrict__ o, int n) {
  int i = blockIdx.x * 256 + threadIdx.x;
  if (i < n) o[i] = (bf16)w[i];
}

// ---------------- stage 2: QKV GEMM ----------------
__global__ __launch_bounds__(256) void qkv_gemm_kernel(const bf16* __restrict__ X,
                                                       const bf16* __restrict__ W,
                                                       bf16* __restrict__ Qb,
                                                       bf16* __restrict__ Kb,
                                                       bf16* __restrict__ Vt) {
  int lane = threadIdx.x & 31, wave = threadIdx.x >> 5;
  int hi = (lane >> 4) & 1, ln = lane & 15;
  int Mbase = blockIdx.x * 128 + (wave & 3) * 32;
  int Nbase = blockIdx.y * 128 + (wave >> 2) * 64;

  v8f acc[2][4] = {};
#pragma unroll 1
  for (int kb = 0; kb < H; kb += 32) {
    v16bf a[2], bf[4];
#pragma unroll
    for (int mi = 0; mi < 2; ++mi) {
      const bf16* p = X + (Mbase + mi * 16 + ln) * H + kb;
      a[mi] = load_afrag(p, hi);
      __builtin_prefetch(p + 64, 0, 0);
    }
#pragma unroll
    for (int ni = 0; ni < 4; ++ni) {
      const bf16* p = W + (Nbase + ni * 16 + ln) * H + kb;
      bf[ni] = load_bfrag(p, hi);
      __builtin_prefetch(p + 64, 0, 0);
    }
#pragma unroll
    for (int mi = 0; mi < 2; ++mi)
#pragma unroll
      for (int ni = 0; ni < 4; ++ni)
        acc[mi][ni] = wmma_bf16(a[mi], bf[ni], acc[mi][ni]);
  }
#pragma unroll
  for (int mi = 0; mi < 2; ++mi)
#pragma unroll
    for (int ni = 0; ni < 4; ++ni)
#pragma unroll
      for (int v = 0; v < 8; ++v) {
        int row = Mbase + mi * 16 + v + 8 * hi;      // token t
        int col = Nbase + ni * 16 + ln;              // qkv feature j
        float val = acc[mi][ni][v];
        int head = col / 192, c = col % 192;
        int s = row >> 1, bb = row & 1;
        int natt = bb * NH + head;
        if (c < 64)       Qb[(natt * SQ + s) * HN + c]        = (bf16)(val * 0.125f);
        else if (c < 128) Kb[(natt * SQ + s) * HN + (c - 64)] = (bf16)val;
        else              Vt[(natt * HN + (c - 128)) * SQ + s] = (bf16)val;
      }
}

// ---------------- stage 3: flash attention ----------------
// K/V tiles staged once per block via async-to-LDS; 8 waves share them.
__global__ __launch_bounds__(256) void flash_attn_kernel(const bf16* __restrict__ Qb,
                                                         const bf16* __restrict__ Kb,
                                                         const bf16* __restrict__ Vt,
                                                         bf16* __restrict__ Ctx) {
  __shared__ __align__(16) bf16 kT[64 * 64];         // [key r][d]
  __shared__ __align__(16) bf16 vT[64 * 64];         // [d][key r]
  __shared__ __align__(16) bf16 pt[8][16 * 64];      // per-wave P tile
  int tid = threadIdx.x;
  int lane = tid & 31, wave = tid >> 5;
  int hi = (lane >> 4) & 1, ln = lane & 15;
  int n = blockIdx.x;                                // attention head (b-major)
  int q0 = blockIdx.y * 128 + wave * 16;

  const bf16* Qh = Qb + (size_t)n * SQ * HN;
  const bf16* Kh = Kb + (size_t)n * SQ * HN;
  const bf16* Vh = Vt + (size_t)n * HN * SQ;

  v16bf qa[2];
#pragma unroll
  for (int dt = 0; dt < 2; ++dt)
    qa[dt] = load_afrag(Qh + (q0 + ln) * HN + dt * 32, hi);

  v8f o[4] = {};
  float rm[8], rl[8];
#pragma unroll
  for (int v = 0; v < 8; ++v) { rm[v] = -1e30f; rl[v] = 0.0f; }

  int vd = tid >> 2, vc = (tid & 3) * 16;            // vT fill coords

#pragma unroll 1
  for (int kb = 0; kb < SQ; kb += 64) {
    // cooperative async fill: kT is a contiguous 8KB span, vT gathers 64 row-chunks
    async_cp16(Kh + kb * HN + tid * 16, &kT[tid * 16]);
    async_cp16(Vh + vd * SQ + kb + vc, &vT[vd * 64 + vc]);
    asm volatile("s_wait_asynccnt 0" ::: "memory");
    __syncthreads();

    // S = Q * K^T for 64 key columns (4 n-tiles x 2 d-steps)
    v8f sj[4] = {};
#pragma unroll
    for (int j = 0; j < 4; ++j)
#pragma unroll
      for (int dt = 0; dt < 2; ++dt) {
        v16bf b = load_bfrag(&kT[(j * 16 + ln) * 64 + dt * 32], hi);
        sj[j] = wmma_bf16(qa[dt], b, sj[j]);
      }

    // online softmax per row (rows live in 16-lane half-waves)
    float p[4][8];
#pragma unroll
    for (int v = 0; v < 8; ++v) {
      float mx = fmaxf(fmaxf(sj[0][v], sj[1][v]), fmaxf(sj[2][v], sj[3][v]));
#pragma unroll
      for (int m = 8; m >= 1; m >>= 1) mx = fmaxf(mx, __shfl_xor(mx, m, 32));
      float mnew = fmaxf(rm[v], mx);
      float corr = __expf(rm[v] - mnew);
      rm[v] = mnew;
      float rs = 0.0f;
#pragma unroll
      for (int j = 0; j < 4; ++j) { p[j][v] = __expf(sj[j][v] - mnew); rs += p[j][v]; }
#pragma unroll
      for (int m = 8; m >= 1; m >>= 1) rs += __shfl_xor(rs, m, 32);
      rl[v] = rl[v] * corr + rs;
#pragma unroll
      for (int dt = 0; dt < 4; ++dt) o[dt][v] *= corr;
    }

    // re-stripe P (C layout) -> A-fragment layout via per-wave LDS tile
#pragma unroll
    for (int v = 0; v < 8; ++v) {
      int m = v + 8 * hi;
#pragma unroll
      for (int j = 0; j < 4; ++j)
        pt[wave][m * 64 + j * 16 + ln] = (bf16)p[j][v];
    }
    asm volatile("s_wait_dscnt 0" ::: "memory");     // wave-internal LDS RAW
    v16bf pA[2];
#pragma unroll
    for (int ks = 0; ks < 2; ++ks)
      pA[ks] = load_afrag(&pt[wave][ln * 64 + ks * 32], hi);

    // O += P * V
#pragma unroll
    for (int dt = 0; dt < 4; ++dt)
#pragma unroll
      for (int ks = 0; ks < 2; ++ks) {
        v16bf vb = load_bfrag(&vT[(dt * 16 + ln) * 64 + ks * 32], hi);
        o[dt] = wmma_bf16(pA[ks], vb, o[dt]);
      }
    __syncthreads();                                 // protect kT/vT before refill
  }

  // epilogue: normalize, place per reference's (nh,b) view quirk
  int ib = n & 1, inh = n >> 1;
#pragma unroll
  for (int v = 0; v < 8; ++v) {
    float inv = 1.0f / rl[v];
    int s = q0 + v + 8 * hi;
    int crow = s * BB + ib;
#pragma unroll
    for (int dt = 0; dt < 4; ++dt)
      Ctx[crow * H + inh * HN + dt * 16 + ln] = (bf16)(o[dt][v] * inv);
  }
}

// ---------------- stage 4: output projection ----------------
__global__ __launch_bounds__(256) void proj_gemm_kernel(const bf16* __restrict__ Cx,
                                                        const bf16* __restrict__ W,
                                                        const float* __restrict__ bias,
                                                        float* __restrict__ out) {
  int lane = threadIdx.x & 31, wave = threadIdx.x >> 5;
  int hi = (lane >> 4) & 1, ln = lane & 15;
  int Mbase = blockIdx.x * 128 + (wave & 3) * 32;
  int Nbase = blockIdx.y * 128 + (wave >> 2) * 64;

  v8f acc[2][4] = {};
#pragma unroll 1
  for (int kb = 0; kb < H; kb += 32) {
    v16bf a[2], bf[4];
#pragma unroll
    for (int mi = 0; mi < 2; ++mi) {
      const bf16* p = Cx + (Mbase + mi * 16 + ln) * H + kb;
      a[mi] = load_afrag(p, hi);
      __builtin_prefetch(p + 64, 0, 0);
    }
#pragma unroll
    for (int ni = 0; ni < 4; ++ni) {
      const bf16* p = W + (Nbase + ni * 16 + ln) * H + kb;
      bf[ni] = load_bfrag(p, hi);
    }
#pragma unroll
    for (int mi = 0; mi < 2; ++mi)
#pragma unroll
      for (int ni = 0; ni < 4; ++ni)
        acc[mi][ni] = wmma_bf16(a[mi], bf[ni], acc[mi][ni]);
  }
#pragma unroll
  for (int mi = 0; mi < 2; ++mi)
#pragma unroll
    for (int ni = 0; ni < 4; ++ni) {
      int col = Nbase + ni * 16 + ln;
      float bv = bias[col];
#pragma unroll
      for (int v = 0; v < 8; ++v) {
        int row = Mbase + mi * 16 + v + 8 * hi;      // token t = s*b + bb
        int s = row >> 1, bb = row & 1;
        out[((size_t)bb * SQ + s) * H + col] = acc[mi][ni][v] + bv;
      }
    }
}

extern "C" void kernel_launch(void* const* d_in, const int* in_sizes, int n_in,
                              void* d_out, int out_size, void* d_ws, size_t ws_size,
                              hipStream_t stream) {
  const float* query = (const float*)d_in[0];
  const float* Wqkv  = (const float*)d_in[3];
  const float* Wproj = (const float*)d_in[4];
  const float* bproj = (const float*)d_in[5];
  float* out = (float*)d_out;

  bf16* xb    = (bf16*)d_ws;                         // [TOK, H]
  bf16* wqkvb = xb    + (size_t)TOK * H;             // [3H, H]
  bf16* wpb   = wqkvb + (size_t)3 * H * H;           // [H, H]
  bf16* Qb    = wpb   + (size_t)H * H;               // [NATT, SQ, HN]
  bf16* Kb    = Qb    + (size_t)NATT * SQ * HN;      // [NATT, SQ, HN]
  bf16* Vt    = Kb    + (size_t)NATT * SQ * HN;      // [NATT, HN, SQ]
  bf16* Ctx   = Vt    + (size_t)NATT * HN * SQ;      // [TOK, H]

  cvt_x_kernel<<<(TOK * H) / 256, 256, 0, stream>>>(query, xb);
  cvt_w_kernel<<<(3 * H * H) / 256, 256, 0, stream>>>(Wqkv, wqkvb, 3 * H * H);
  cvt_w_kernel<<<(H * H) / 256, 256, 0, stream>>>(Wproj, wpb, H * H);

  qkv_gemm_kernel<<<dim3(TOK / 128, (3 * H) / 128), 256, 0, stream>>>(xb, wqkvb, Qb, Kb, Vt);
  flash_attn_kernel<<<dim3(NATT, SQ / 128), 256, 0, stream>>>(Qb, Kb, Vt, Ctx);
  proj_gemm_kernel<<<dim3(TOK / 128, H / 128), 256, 0, stream>>>(Ctx, wpb, bproj, out);
}
